// TGCNCell_49572512530454
// MI455X (gfx1250) — matrix operand order repaired and checked by reading, not compiled, not run on hardware
//
#include <hip/hip_runtime.h>
#include <cstdint>

// ---------------------------------------------------------------------------
// TGCN cell with wind-based dynamic adjacency, CBAM attention, GRU update.
// MI455X / gfx1250: all GEMMs via v_wmma_f32_16x16x32_f16 (wave32, 16x16 tiles).
// - a1 (normalized Laplacian) is hidden-independent: computed ONCE per batch,
//   reused for both graph convs (reference computes the 2000^3 GEMM twice).
// - Big GEMMs use 4x2 register tile-blocking per wave: 12 b128 loads feed
//   8 WMMAs per K-step (~21 FLOP/byte of cache traffic vs 8 for 1x1).
// - M/N padded to 2048 (zero-filled) so 64x32 wave tiles divide evenly;
//   epilogue stores guarded to real dimensions.
// ---------------------------------------------------------------------------

#define B_   8
#define N_   2000
#define H_   64
#define KP_  2016          // K padded to multiple of 32 (2016 = 63*32)
#define MP_  2048          // M/N padded to multiple of 64 for 4x2 wave tiles
#define NB_  80            // (H+1)=65 output columns padded to 5 tiles of 16
#define C2H_ 128           // 2*H
#define NH_  (N_*H_)

typedef _Float16 half_t;
typedef __attribute__((ext_vector_type(16))) _Float16 v16h;
typedef __attribute__((ext_vector_type(8)))  float    v8f;

union Frag16 { uint32_t u[8]; v16h v; };

__device__ __forceinline__ float sigmoidf_(float x) { return 1.f / (1.f + expf(-x)); }

// ---------------------------------------------------------------------------
// NT GEMM: C[M x Nc] = A[M x K] * Bt[Nc x K]^T, f16 in / f32 accumulate.
// Wave computes a (BM*16) x (BN*16) block of C: BM*BN accumulators, BM+BN
// fragments per K-step, BM*BN WMMAs per K-step.
// Fragment layouts per CDNA5 ISA 7.12.2 (wave32):
//   A 16x32 f16 : lane&15 = row M; lanes>=16 get K offset +8;
//                 VGPR0..3 = K {0..7}, VGPR4..7 = K {16..23} (pairs packed).
//   B 32x16 f16 : lane&15 = column N; same K layout as A (lanes>=16 +8);
//                 VGPR0..3 = K {0..7}, VGPR4..7 = K {16..23} (pairs packed).
//   C/D 16x16 f32: VGPR r -> M = r + 8*(lane>=16), N = lane&15.
// MODE 0: C[m,n] = acc (+ biasN[n])                               (f32 out)
// MODE 1: Obf[m,n] = f16( relu(acc + biasN[n] + Smat[m,n]) + (m==n) )
// ---------------------------------------------------------------------------
template<int BM, int BN, int MODE>
__global__ __launch_bounds__(256) void wmma_gemm_nt(
    const half_t* __restrict__ A, int lda,
    const half_t* __restrict__ Bt, int ldb,
    int Mreal, int Nreal, int K, int tmBlocks, int tnBlocks,
    float* __restrict__ Cf, int ldc, const float* __restrict__ biasN,
    half_t* __restrict__ Obf, int ldo,
    const float* __restrict__ Smat, int ldS)
{
    int wave  = blockIdx.x * (blockDim.x >> 5) + (threadIdx.x >> 5);
    int total = tmBlocks * tnBlocks;
    if (wave >= total) return;                 // wave-uniform: EXEC stays all-1s
    int bm = wave / tnBlocks;
    int bn = wave - bm * tnBlocks;
    int lane = threadIdx.x & 31;
    int hi   = lane >> 4;
    int lr   = lane & 15;

    const half_t* arow[BM];
    const half_t* brow[BN];
#pragma unroll
    for (int i = 0; i < BM; ++i)
        arow[i] = A + (size_t)(bm * (BM * 16) + i * 16 + lr) * lda + (hi << 3);
#pragma unroll
    for (int j = 0; j < BN; ++j)
        brow[j] = Bt + (size_t)(bn * (BN * 16) + j * 16 + lr) * ldb + (hi << 3);

    v8f acc[BM][BN] = {};
    for (int k0 = 0; k0 < K; k0 += 32) {
        Frag16 af[BM], bf[BN];
#pragma unroll
        for (int i = 0; i < BM; ++i) {
            const uint32_t* ap = (const uint32_t*)(arow[i] + k0);
            af[i].u[0] = ap[0];  af[i].u[1] = ap[1];
            af[i].u[2] = ap[2];  af[i].u[3] = ap[3];
            af[i].u[4] = ap[8];  af[i].u[5] = ap[9];
            af[i].u[6] = ap[10]; af[i].u[7] = ap[11];
        }
#pragma unroll
        for (int j = 0; j < BN; ++j) {
            const uint32_t* bp = (const uint32_t*)(brow[j] + k0);
#pragma unroll
            for (int v = 0; v < 8; ++v) bf[j].u[v] = bp[(v & 3) + ((v >> 2) << 3)];
        }
#pragma unroll
        for (int i = 0; i < BM; ++i)
#pragma unroll
            for (int j = 0; j < BN; ++j)
                acc[i][j] = __builtin_amdgcn_wmma_f32_16x16x32_f16(
                    false, af[i].v, false, bf[j].v, (short)0, acc[i][j],
                    false, false);
    }

#pragma unroll
    for (int i = 0; i < BM; ++i) {
#pragma unroll
        for (int j = 0; j < BN; ++j) {
            int n = bn * (BN * 16) + j * 16 + lr;
#pragma unroll
            for (int r = 0; r < 8; ++r) {
                int m = bm * (BM * 16) + i * 16 + r + (hi << 3);
                if (m < Mreal && n < Nreal) {
                    float val = acc[i][j][r];
                    if (MODE == 0) {
                        if (biasN) val += biasN[n];
                        Cf[(size_t)m * ldc + n] = val;
                    } else {
                        val += biasN[n] + Smat[(size_t)m * ldS + n];
                        val = fmaxf(val, 0.f) + ((m == n) ? 1.f : 0.f);
                        Obf[(size_t)m * ldo + n] = (half_t)val;
                    }
                }
            }
        }
    }
}

// --------------------------- support kernels --------------------------------

// f32 (N_ x N_) -> f16 (rows x KP_), zero-padded tail rows/columns.
__global__ void k_f32_to_f16_pad(const float* __restrict__ src,
                                 half_t* __restrict__ dst, int rows)
{
    int idx = blockIdx.x * blockDim.x + threadIdx.x;
    if (idx >= rows * KP_) return;
    int i = idx / KP_, j = idx - i * KP_;
    dst[idx] = (i < N_ && j < N_) ? (half_t)src[i * N_ + j] : (half_t)0.f;
}

// d1[b,i,j] = max(cos(clip(|windd[i,j] - 360*inputs[b,2,i]|) * pi/180), 0)
__global__ void k_build_d1(const float* __restrict__ inp_b,   // inputs + b*3N
                           const float* __restrict__ windd,
                           half_t* __restrict__ d1)
{
    int idx = blockIdx.x * blockDim.x + threadIdx.x;
    if (idx >= MP_ * KP_) return;
    int i = idx / KP_, j = idx - i * KP_;
    half_t v = (half_t)0.f;
    if (i < N_ && j < N_) {
        float wn = inp_b[2 * N_ + i] * 360.f;
        float dr = fabsf(windd[i * N_ + j] - wn);
        if (dr > 360.f) dr = 180.f;
        v = (half_t)fmaxf(cosf(dr * 0.017453292519943295f), 0.f);
    }
    d1[idx] = v;
}

// dinv[i] = rsqrt(sum_j m[i,j])   (m already contains relu(a)+I, sum >= 1)
__global__ __launch_bounds__(256) void k_rowsum_rsqrt(
    const half_t* __restrict__ m, float* __restrict__ dinv)
{
    __shared__ float red[256];
    int i = blockIdx.x;
    const half_t* row = m + (size_t)i * KP_;
    float s = 0.f;
    for (int j = threadIdx.x; j < N_; j += 256) s += (float)row[j];
    red[threadIdx.x] = s;
    __syncthreads();
    for (int st = 128; st > 0; st >>= 1) {
        if (threadIdx.x < st) red[threadIdx.x] += red[threadIdx.x + st];
        __syncthreads();
    }
    if (threadIdx.x == 0) dinv[i] = rsqrtf(red[0]);
}

// a1[i,j] = dinv[i] * m[i,j] * dinv[j]   (f16 out, zero-padded rows/cols)
__global__ void k_normalize(const half_t* __restrict__ m,
                            const float* __restrict__ dinv,
                            half_t* __restrict__ a1)
{
    int idx = blockIdx.x * blockDim.x + threadIdx.x;
    if (idx >= MP_ * KP_) return;
    int i = idx / KP_, j = idx - i * KP_;
    half_t v = (half_t)0.f;
    if (i < N_ && j < N_)
        v = (half_t)(dinv[i] * (float)m[idx] * dinv[j]);
    a1[idx] = v;
}

// xT[h', j] : row 0 = inputs[b,0,j], rows 1..64 = hidden[b, j*H + h'-1], rest 0.
__global__ void k_build_x1T(const float* __restrict__ inp_b,
                            const float* __restrict__ hid_b,
                            half_t* __restrict__ xT)
{
    int idx = blockIdx.x * blockDim.x + threadIdx.x;
    if (idx >= NB_ * KP_) return;
    int hp = idx / KP_, j = idx - hp * KP_;
    float v = 0.f;
    if (j < N_) {
        if (hp == 0)          v = inp_b[j];
        else if (hp <= H_)    v = hid_b[j * H_ + (hp - 1)];
    }
    xT[idx] = (half_t)v;
}

// same, but hidden gated by r (flat elementwise, r = gate[0 : N*H])
__global__ void k_build_x2T(const float* __restrict__ inp_b,
                            const float* __restrict__ hid_b,
                            const float* __restrict__ gate,
                            half_t* __restrict__ xT)
{
    int idx = blockIdx.x * blockDim.x + threadIdx.x;
    if (idx >= NB_ * KP_) return;
    int hp = idx / KP_, j = idx - hp * KP_;
    float v = 0.f;
    if (j < N_) {
        if (hp == 0)          v = inp_b[j];
        else if (hp <= H_) {
            int k = j * H_ + (hp - 1);
            v = gate[k] * hid_b[k];
        }
    }
    xT[idx] = (half_t)v;
}

// ybuf[i*128 + c] = b1a[c] + sum_h y1[i, h] * W1a[c, h]     (h over 0..64)
// Note: linearized ybuf IS the (B,2H,N)-reshaped conc: conc[c,j] = ybuf[c*N+j].
__global__ void k_w1a(const float* __restrict__ y1,
                      const float* __restrict__ W1a,
                      const float* __restrict__ b1a,
                      float* __restrict__ ybuf)
{
    int idx = blockIdx.x * blockDim.x + threadIdx.x;
    if (idx >= N_ * C2H_) return;
    int i = idx / C2H_, c = idx - i * C2H_;
    float acc = b1a[c];
    const float* yr = y1 + (size_t)i * NB_;
    const float* wr = W1a + c * (H_ + 1);
    for (int h = 0; h <= H_; ++h) acc += yr[h] * wr[h];
    ybuf[idx] = acc;
}

// per-channel mean/max over N (CBAM channel attention input)
__global__ __launch_bounds__(256) void k_ca_reduce(
    const float* __restrict__ ybuf, float* __restrict__ camean,
    float* __restrict__ camax)
{
    __shared__ float rs[256], rm[256];
    int c = blockIdx.x;
    const float* p = ybuf + (size_t)c * N_;
    float s = 0.f, mx = -3.4e38f;
    for (int j = threadIdx.x; j < N_; j += 256) {
        float v = p[j]; s += v; mx = fmaxf(mx, v);
    }
    rs[threadIdx.x] = s; rm[threadIdx.x] = mx;
    __syncthreads();
    for (int st = 128; st > 0; st >>= 1) {
        if (threadIdx.x < st) {
            rs[threadIdx.x] += rs[threadIdx.x + st];
            rm[threadIdx.x]  = fmaxf(rm[threadIdx.x], rm[threadIdx.x + st]);
        }
        __syncthreads();
    }
    if (threadIdx.x == 0) { camean[c] = rs[0] / (float)N_; camax[c] = rm[0]; }
}

// ca[c] = sigmoid( Wca2 @ (relu(Wca1@avg) + relu(Wca1@max)) )   (1 block, 128 thr)
__global__ void k_ca_fc(const float* __restrict__ camean,
                        const float* __restrict__ camax,
                        const float* __restrict__ Wca1,
                        const float* __restrict__ Wca2,
                        float* __restrict__ ca)
{
    __shared__ float hid[8];
    int tid = threadIdx.x;
    if (tid < 8) {
        float sa = 0.f, sm = 0.f;
        const float* w = Wca1 + tid * C2H_;
        for (int c = 0; c < C2H_; ++c) { sa += w[c] * camean[c]; sm += w[c] * camax[c]; }
        hid[tid] = fmaxf(sa, 0.f) + fmaxf(sm, 0.f);
    }
    __syncthreads();
    float o = 0.f;
    const float* w2 = Wca2 + tid * 8;
    for (int k = 0; k < 8; ++k) o += w2[k] * hid[k];
    ca[tid] = sigmoidf_(o);
}

// spatial attention input: mean/max over channels of (ca[c] * conc[c,j])
__global__ void k_sa_reduce(const float* __restrict__ ybuf,
                            const float* __restrict__ ca,
                            float* __restrict__ s0, float* __restrict__ s1)
{
    int j = blockIdx.x * blockDim.x + threadIdx.x;
    if (j >= N_) return;
    float s = 0.f, mx = -3.4e38f;
    for (int c = 0; c < C2H_; ++c) {
        float v = ca[c] * ybuf[(size_t)c * N_ + j];
        s += v; mx = fmaxf(mx, v);
    }
    s0[j] = s / (float)C2H_;
    s1[j] = mx;
}

// width-7 conv over j (padding 3), then sigmoid
__global__ void k_sa_conv(const float* __restrict__ s0,
                          const float* __restrict__ s1,
                          const float* __restrict__ Wsa,
                          float* __restrict__ sigsa)
{
    int j = blockIdx.x * blockDim.x + threadIdx.x;
    if (j >= N_) return;
    float a = 0.f;
    for (int t = 0; t < 7; ++t) {
        int jj = j + t - 3;
        if (jj >= 0 && jj < N_) a += Wsa[t] * s0[jj] + Wsa[7 + t] * s1[jj];
    }
    sigsa[j] = sigmoidf_(a);
}

// gate[f] = sigmoid( ca[f/N] * sigsa[f%N] * conc_raw[f] ); r = gate[:NH], u = gate[NH:]
__global__ void k_gate(const float* __restrict__ ybuf,
                       const float* __restrict__ ca,
                       const float* __restrict__ sigsa,
                       float* __restrict__ gate)
{
    int f = blockIdx.x * blockDim.x + threadIdx.x;
    if (f >= C2H_ * N_) return;
    int c = f / N_, j = f - c * N_;
    gate[f] = sigmoidf_(ca[c] * sigsa[j] * ybuf[f]);
}

// c = tanh(y2 @ W1b^T + b1b); new_h = u*h + (1-u)*c
__global__ void k_final(const float* __restrict__ y2,
                        const float* __restrict__ W1b,
                        const float* __restrict__ b1b,
                        const float* __restrict__ gate,
                        const float* __restrict__ hid_b,
                        float* __restrict__ out_b)
{
    int idx = blockIdx.x * blockDim.x + threadIdx.x;
    if (idx >= NH_) return;
    int j = idx / H_, h = idx - j * H_;
    float acc = b1b[h];
    const float* yr = y2 + (size_t)j * NB_;
    const float* wr = W1b + h * (H_ + 1);
    for (int t = 0; t <= H_; ++t) acc += yr[t] * wr[t];
    float cval = tanhf(acc);
    float u = gate[NH_ + idx];
    out_b[idx] = u * hid_b[idx] + (1.f - u) * cval;
}

// ---------------------------------------------------------------------------

extern "C" void kernel_launch(void* const* d_in, const int* in_sizes, int n_in,
                              void* d_out, int out_size, void* d_ws, size_t ws_size,
                              hipStream_t stream)
{
    (void)in_sizes; (void)n_in; (void)out_size; (void)ws_size;

    const float* inputs = (const float*)d_in[0];
    const float* hidden = (const float*)d_in[1];
    const float* adj    = (const float*)d_in[2];
    const float* windd  = (const float*)d_in[3];
    const float* W2     = (const float*)d_in[4];
    const float* b2     = (const float*)d_in[5];
    const float* W3     = (const float*)d_in[6];
    const float* b3     = (const float*)d_in[7];
    const float* W1a    = (const float*)d_in[8];
    const float* b1a    = (const float*)d_in[9];
    const float* W1b    = (const float*)d_in[10];
    const float* b1b    = (const float*)d_in[11];
    const float* Wca1   = (const float*)d_in[12];
    const float* Wca2   = (const float*)d_in[13];
    const float* Wsa    = (const float*)d_in[14];
    float* out = (float*)d_out;

    // ---- workspace carve-out (~68 MB total, 256B-aligned blocks) ----
    char* base = (char*)d_ws;
    size_t off = 0;
    auto take = [&](size_t bytes) -> char* {
        char* p = base + off;
        off = (off + bytes + 255) & ~(size_t)255;
        return p;
    };
    half_t* W2b   = (half_t*)take((size_t)MP_ * KP_ * 2);
    half_t* W3b   = (half_t*)take((size_t)MP_ * KP_ * 2);
    half_t* adjb  = (half_t*)take((size_t)MP_ * KP_ * 2);
    float*  Smat  = (float*) take((size_t)N_ * N_ * 4);
    half_t* d1b   = (half_t*)take((size_t)MP_ * KP_ * 2);
    half_t* mb    = (half_t*)take((size_t)N_ * KP_ * 2);
    half_t* a1b   = (half_t*)take((size_t)MP_ * KP_ * 2);
    float*  dinv  = (float*) take((size_t)N_ * 4);
    half_t* x1T   = (half_t*)take((size_t)NB_ * KP_ * 2);
    half_t* x2T   = (half_t*)take((size_t)NB_ * KP_ * 2);
    float*  y1    = (float*) take((size_t)N_ * NB_ * 4);
    float*  y2    = (float*) take((size_t)N_ * NB_ * 4);
    float*  ybuf  = (float*) take((size_t)N_ * C2H_ * 4);
    float*  gate  = (float*) take((size_t)C2H_ * N_ * 4);
    float*  camean= (float*) take(C2H_ * 4);
    float*  camax = (float*) take(C2H_ * 4);
    float*  cav   = (float*) take(C2H_ * 4);
    float*  s0    = (float*) take(N_ * 4);
    float*  s1    = (float*) take(N_ * 4);
    float*  sigsa = (float*) take(N_ * 4);

    auto cdiv = [](int a, int b) { return (a + b - 1) / b; };
    const int T = 256;
    const int gMPK = cdiv(MP_ * KP_, T);

    // big GEMMs: 4x2 wave tiles over padded 2048x2048 output
    const int tmBig = MP_ / 64;       // 32
    const int tnBig = MP_ / 32;       // 64
    const int wavesBig = tmBig * tnBig;   // 2048 waves
    // y GEMMs: 1x1 wave tiles, real M (2000/16=125), Nc=80 (5 tiles)
    const int tmY = N_ / 16, tnY = NB_ / 16;
    const int wavesY = tmY * tnY;         // 625 waves

    // ---- batch-independent precompute ----
    k_f32_to_f16_pad<<<gMPK, T, 0, stream>>>(W2,  W2b,  MP_);
    k_f32_to_f16_pad<<<gMPK, T, 0, stream>>>(W3,  W3b,  MP_);
    k_f32_to_f16_pad<<<gMPK, T, 0, stream>>>(adj, adjb, MP_);
    // S = adj @ W3^T + b3
    wmma_gemm_nt<4, 2, 0><<<cdiv(wavesBig, 8), T, 0, stream>>>(
        adjb, KP_, W3b, KP_, N_, N_, KP_, tmBig, tnBig,
        Smat, N_, b3, nullptr, 0, nullptr, 0);

    // ---- per-batch pipeline ----
    for (int b = 0; b < B_; ++b) {
        const float* inp_b = inputs + (size_t)b * 3 * N_;
        const float* hid_b = hidden + (size_t)b * NH_;
        float*       out_b = out    + (size_t)b * NH_;

        // dynamic adjacency and normalized Laplacian a1 (computed ONCE per batch)
        k_build_d1<<<gMPK, T, 0, stream>>>(inp_b, windd, d1b);
        // m = relu(d1 @ W2^T + b2 + S) + I   (fused epilogue, f16 out)
        wmma_gemm_nt<4, 2, 1><<<cdiv(wavesBig, 8), T, 0, stream>>>(
            d1b, KP_, W2b, KP_, N_, N_, KP_, tmBig, tnBig,
            nullptr, 0, b2, mb, KP_, Smat, N_);
        k_rowsum_rsqrt<<<N_, T, 0, stream>>>(mb, dinv);
        k_normalize<<<gMPK, T, 0, stream>>>(mb, dinv, a1b);

        // graph conv 1: y1 = a1 @ [x | h]
        k_build_x1T<<<cdiv(NB_ * KP_, T), T, 0, stream>>>(inp_b, hid_b, x1T);
        wmma_gemm_nt<1, 1, 0><<<cdiv(wavesY, 8), T, 0, stream>>>(
            a1b, KP_, x1T, KP_, N_, NB_, KP_, tmY, tnY,
            y1, NB_, nullptr, nullptr, 0, nullptr, 0);
        k_w1a<<<cdiv(N_ * C2H_, T), T, 0, stream>>>(y1, W1a, b1a, ybuf);

        // CBAM channel + spatial attention, GRU gate sigmoid
        k_ca_reduce<<<C2H_, T, 0, stream>>>(ybuf, camean, camax);
        k_ca_fc<<<1, C2H_, 0, stream>>>(camean, camax, Wca1, Wca2, cav);
        k_sa_reduce<<<cdiv(N_, T), T, 0, stream>>>(ybuf, cav, s0, s1);
        k_sa_conv<<<cdiv(N_, T), T, 0, stream>>>(s0, s1, Wsa, sigsa);
        k_gate<<<cdiv(C2H_ * N_, T), T, 0, stream>>>(ybuf, cav, sigsa, gate);

        // graph conv 2 (reusing a1): y2 = a1 @ [x | r*h]
        k_build_x2T<<<cdiv(NB_ * KP_, T), T, 0, stream>>>(inp_b, hid_b, gate, x2T);
        wmma_gemm_nt<1, 1, 0><<<cdiv(wavesY, 8), T, 0, stream>>>(
            a1b, KP_, x2T, KP_, N_, NB_, KP_, tmY, tnY,
            y2, NB_, nullptr, nullptr, 0, nullptr, 0);

        // c = tanh(y2 @ W1b^T + b1b); new_h = u*h + (1-u)*c
        k_final<<<cdiv(NH_, T), T, 0, stream>>>(y2, W1b, b1b, gate, hid_b, out_b);
    }
}
